// MDMA_38671885533698
// MI455X (gfx1250) — matrix operand order repaired and checked by reading, not compiled, not run on hardware
//
#include <hip/hip_runtime.h>
#include <hip/hip_bf16.h>

#define DD 64
#define MM 256
#define NNB 1024

typedef __attribute__((ext_vector_type(16))) __bf16 v16bf;
typedef __attribute__((ext_vector_type(8)))  float  v8f;

// softplus(10x)/10, numerically stable
__device__ __forceinline__ float sp10f(float x) {
    float t = 10.0f * x;
    float r;
    if (t > 20.0f)       r = t;
    else if (t < -20.0f) r = expf(t);
    else                 r = logf(1.0f + expf(t));
    return 0.1f * r;
}

__device__ __forceinline__ float blockReduceSum256(float v, float* sm) {
    int t = threadIdx.x;
    sm[t] = v;
    __syncthreads();
    for (int s = 128; s > 0; s >>= 1) {
        if (t < s) sm[t] += sm[t + s];
        __syncthreads();
    }
    float r = sm[0];
    __syncthreads();
    return r;
}

// ---------------------------------------------------------------------------
// Phase 1: per-(n,d,m) rank-3 MLP chain -> phidot, written as T0[n][d][m]
// Block: 256 threads (thread = m), blockIdx.x = d, blockIdx.y = n-tile of 16.
// Params for (d,m) loaded once, reused across 16 n values.
// ---------------------------------------------------------------------------
__global__ void __launch_bounds__(256) mlp_chain_kernel(
    const float* __restrict__ X,        // (N, D)
    const float* __restrict__ w_first,  // (D, M, 1, 3)
    const float* __restrict__ w_mid,    // (2, D, M, 3, 3)
    const float* __restrict__ w_last,   // (D, M, 3, 1)
    const float* __restrict__ b_first,  // (D, M, 3)
    const float* __restrict__ b_mid,    // (2, D, M, 3)
    const float* __restrict__ b_last,   // (D, M, 1)
    const float* __restrict__ a_first,  // (D, M, 3)
    const float* __restrict__ a_mid,    // (2, D, M, 3)
    float* __restrict__ T0)             // (N, 64, 256)
{
    const int d = blockIdx.x;
    const int m = threadIdx.x;
    const int nbase = blockIdx.y * 16;
    const int dm = d * MM + m;

    float W1[3], B1[3], TA1[3], WL[3];
    float Wm[2][3][3], Bm[2][3], TAm[2][3];

#pragma unroll
    for (int l = 0; l < 3; ++l) {
        W1[l]  = sp10f(w_first[dm * 3 + l]);
        B1[l]  = b_first[dm * 3 + l];
        TA1[l] = tanhf(a_first[dm * 3 + l]);
        WL[l]  = sp10f(w_last[dm * 3 + l]);
    }
#pragma unroll
    for (int i = 0; i < 2; ++i) {
        const size_t base9 = ((size_t)(i * DD + d) * MM + m) * 9;
        const size_t base3 = ((size_t)(i * DD + d) * MM + m) * 3;
#pragma unroll
        for (int k = 0; k < 3; ++k)
#pragma unroll
            for (int l = 0; l < 3; ++l)
                Wm[i][k][l] = sp10f(w_mid[base9 + k * 3 + l]);
#pragma unroll
        for (int l = 0; l < 3; ++l) {
            Bm[i][l]  = b_mid[base3 + l];
            TAm[i][l] = tanhf(a_mid[base3 + l]);
        }
    }
    const float BL = b_last[dm];

    for (int nn = 0; nn < 16; ++nn) {
        const int n = nbase + nn;
        const float x = X[(size_t)n * DD + d];

        float phi[3], pd[3];
#pragma unroll
        for (int l = 0; l < 3; ++l) {
            float z  = x * W1[l] + B1[l];
            float tz = tanhf(z);
            pd[l]  = W1[l] * (1.0f + TA1[l] * (1.0f - tz * tz));
            phi[l] = z + tz * TA1[l];
        }
#pragma unroll
        for (int i = 0; i < 2; ++i) {
            float nphi[3], npd[3];
#pragma unroll
            for (int l = 0; l < 3; ++l) {
                float z = Bm[i][l];
                float s = 0.0f;
#pragma unroll
                for (int k = 0; k < 3; ++k) {
                    z += phi[k] * Wm[i][k][l];
                    s += pd[k] * Wm[i][k][l];
                }
                float tz = tanhf(z);
                npd[l]  = s * (1.0f + TAm[i][l] * (1.0f - tz * tz));
                nphi[l] = z + tz * TAm[i][l];
            }
#pragma unroll
            for (int l = 0; l < 3; ++l) { phi[l] = nphi[l]; pd[l] = npd[l]; }
        }
        float z = BL, s = 0.0f;
#pragma unroll
        for (int k = 0; k < 3; ++k) { z += phi[k] * WL[k]; s += pd[k] * WL[k]; }
        float sg = 1.0f / (1.0f + expf(-z));
        T0[((size_t)n * DD + d) * MM + m] = s * sg * (1.0f - sg);
    }
}

// ---------------------------------------------------------------------------
// Prep: Anorm^T in bf16.  A = sp10(a); A /= sum over m (axis=1);
// stored transposed: AnT[g][k][m]  (so WMMA B-fragment loads are contiguous)
// grid = (G, 256), block = 256 (thread = m)
// ---------------------------------------------------------------------------
__global__ void __launch_bounds__(256) anorm_kernel(
    const float* __restrict__ a,   // (G, M, M)
    __bf16* __restrict__ AnT)      // (G, M(k), M(m))
{
    __shared__ float sm[256];
    const int g = blockIdx.x;
    const int k = blockIdx.y;
    const int m = threadIdx.x;
    float v = sp10f(a[((size_t)g * MM + m) * MM + k]);
    float s = blockReduceSum256(v, sm);
    AnT[((size_t)g * MM + k) * MM + m] = (__bf16)(v / s);
}

__global__ void __launch_bounds__(256) a5norm_kernel(
    const float* __restrict__ a5,  // (1, M, 1)
    float* __restrict__ A5n)       // (M)
{
    __shared__ float sm[256];
    const int m = threadIdx.x;
    float v = sp10f(a5[m]);
    float s = blockReduceSum256(v, sm);
    A5n[m] = v / s;
}

// ---------------------------------------------------------------------------
// HT level: Tin (N, 2G, M) -> pair product -> GEMM with AnT[g] -> Tout (N,G,M)
// Tout[n][g][k] = sum_m Tin[n][2g][m]*Tin[n][2g+1][m] * Anorm[g][m][k]
// One wave computes a 16(n) x 128(k) tile: the (computed) A fragment is
// reused across 8 WMMAs per K-chunk.  Block = 8 waves -> 128 n.
// grid = (G, N/128, 256/128).  bf16 WMMA, f32 accumulate over M=256.
// ---------------------------------------------------------------------------
__global__ void __launch_bounds__(256) ht_level_kernel(
    const float*  __restrict__ Tin,
    float*        __restrict__ Tout,
    const __bf16* __restrict__ AnT,
    int G)
{
    const int lane = threadIdx.x & 31;
    const int wid  = threadIdx.x >> 5;
    const int half = lane >> 4;     // 0: lanes 0-15, 1: lanes 16-31
    const int lq   = lane & 15;

    const int g  = blockIdx.x;
    const int n0 = blockIdx.y * 128 + wid * 16;
    const int k0 = blockIdx.z * 128;

    v8f acc[8] = {};

    // A-matrix row for this lane (row = n0+lq), pair rows 2g / 2g+1
    const float* rowA = Tin + ((size_t)(n0 + lq) * (2 * G) + 2 * g) * MM;
    const float* rowB = rowA + MM;

    for (int mb = 0; mb < MM; mb += 32) {
        // Build 16x32 bf16 A fragment (contiguous K, matching B):
        //  lane<16 : elems 0..15 -> K=0..15
        //  lane>=16: elems 0..15 -> K=16..31
        v16bf af;
        {
            const int m1 = mb + 16 * half;
            float4 x0 = *(const float4*)(rowA + m1);
            float4 x1 = *(const float4*)(rowA + m1 + 4);
            float4 y0 = *(const float4*)(rowB + m1);
            float4 y1 = *(const float4*)(rowB + m1 + 4);
            af[0] = (__bf16)(x0.x * y0.x); af[1] = (__bf16)(x0.y * y0.y);
            af[2] = (__bf16)(x0.z * y0.z); af[3] = (__bf16)(x0.w * y0.w);
            af[4] = (__bf16)(x1.x * y1.x); af[5] = (__bf16)(x1.y * y1.y);
            af[6] = (__bf16)(x1.z * y1.z); af[7] = (__bf16)(x1.w * y1.w);
            const int m2 = mb + 16 * half + 8;
            x0 = *(const float4*)(rowA + m2);
            x1 = *(const float4*)(rowA + m2 + 4);
            y0 = *(const float4*)(rowB + m2);
            y1 = *(const float4*)(rowB + m2 + 4);
            af[8]  = (__bf16)(x0.x * y0.x); af[9]  = (__bf16)(x0.y * y0.y);
            af[10] = (__bf16)(x0.z * y0.z); af[11] = (__bf16)(x0.w * y0.w);
            af[12] = (__bf16)(x1.x * y1.x); af[13] = (__bf16)(x1.y * y1.y);
            af[14] = (__bf16)(x1.z * y1.z); af[15] = (__bf16)(x1.w * y1.w);
        }

#pragma unroll
        for (int t = 0; t < 8; ++t) {
            // 32x16 bf16 B fragment: lane's column = k0+t*16+lq,
            // K range 16*half..16*half+15 -> contiguous m in AnT[g][k][m]
            const int kc = k0 + t * 16 + lq;
            const __bf16* bp = AnT + ((size_t)g * MM + kc) * MM + mb + 16 * half;
            union { uint4 u[2]; v16bf v; } bf_;
            bf_.u[0] = ((const uint4*)bp)[0];
            bf_.u[1] = ((const uint4*)bp)[1];
            acc[t] = __builtin_amdgcn_wmma_f32_16x16x32_bf16(
                false, af, false, bf_.v, (short)0, acc[t], false, false);
        }
    }

    // D layout: lane's column = lq, VGPR v -> row v + 8*half
#pragma unroll
    for (int t = 0; t < 8; ++t) {
#pragma unroll
        for (int v = 0; v < 8; ++v) {
            const int n = n0 + v + 8 * half;
            const int k = k0 + t * 16 + lq;
            Tout[((size_t)n * G + g) * MM + k] = acc[t][v];
        }
    }
}

// ---------------------------------------------------------------------------
// Final: Tin (N,2,256) -> out[n] = sum_m Tin[n][0][m]*Tin[n][1][m]*A5n[m]
// ---------------------------------------------------------------------------
__global__ void __launch_bounds__(256) final_kernel(
    const float* __restrict__ Tin,
    const float* __restrict__ A5n,
    float* __restrict__ out)
{
    __shared__ float sm[256];
    const int n = blockIdx.x;
    const int m = threadIdx.x;
    float p = Tin[((size_t)n * 2 + 0) * MM + m] *
              Tin[((size_t)n * 2 + 1) * MM + m] * A5n[m];
    float s = blockReduceSum256(p, sm);
    if (m == 0) out[n] = s;
}

extern "C" void kernel_launch(void* const* d_in, const int* in_sizes, int n_in,
                              void* d_out, int out_size, void* d_ws, size_t ws_size,
                              hipStream_t stream) {
    (void)in_sizes; (void)n_in; (void)out_size; (void)ws_size;

    const float* X       = (const float*)d_in[0];
    const float* w_first = (const float*)d_in[1];
    const float* w_mid   = (const float*)d_in[2];
    const float* w_last  = (const float*)d_in[3];
    const float* b_first = (const float*)d_in[4];
    const float* b_mid   = (const float*)d_in[5];
    const float* b_last  = (const float*)d_in[6];
    const float* a_first = (const float*)d_in[7];
    const float* a_mid   = (const float*)d_in[8];
    const float* aHT[5]  = {(const float*)d_in[9],  (const float*)d_in[10],
                            (const float*)d_in[11], (const float*)d_in[12],
                            (const float*)d_in[13]};
    const float* aHT5    = (const float*)d_in[14];
    float* out = (float*)d_out;

    char* ws = (char*)d_ws;
    size_t off = 0;
    float* T0 = (float*)(ws + off); off += (size_t)NNB * 64 * MM * sizeof(float);
    float* T1 = (float*)(ws + off); off += (size_t)NNB * 32 * MM * sizeof(float);
    const int Gs[5] = {32, 16, 8, 4, 2};
    __bf16* AnT[5];
    for (int i = 0; i < 5; ++i) {
        AnT[i] = (__bf16*)(ws + off);
        off += (size_t)Gs[i] * MM * MM * sizeof(__bf16);
    }
    float* A5n = (float*)(ws + off); off += MM * sizeof(float);

    // Phase 1: MLP chains -> T0 (N,64,256)
    mlp_chain_kernel<<<dim3(DD, NNB / 16), 256, 0, stream>>>(
        X, w_first, w_mid, w_last, b_first, b_mid, b_last, a_first, a_mid, T0);

    // Prep normalized mixing matrices (bf16, transposed)
    for (int i = 0; i < 5; ++i)
        anorm_kernel<<<dim3(Gs[i], MM), 256, 0, stream>>>(aHT[i], AnT[i]);
    a5norm_kernel<<<1, 256, 0, stream>>>(aHT5, A5n);

    // HT levels (ping-pong T0 <-> T1)
    ht_level_kernel<<<dim3(32, NNB / 128, 2), 256, 0, stream>>>(T0, T1, AnT[0], 32);
    ht_level_kernel<<<dim3(16, NNB / 128, 2), 256, 0, stream>>>(T1, T0, AnT[1], 16);
    ht_level_kernel<<<dim3( 8, NNB / 128, 2), 256, 0, stream>>>(T0, T1, AnT[2],  8);
    ht_level_kernel<<<dim3( 4, NNB / 128, 2), 256, 0, stream>>>(T1, T0, AnT[3],  4);
    ht_level_kernel<<<dim3( 2, NNB / 128, 2), 256, 0, stream>>>(T0, T1, AnT[4],  2);

    // Final contraction with aHT5
    final_kernel<<<NNB, 256, 0, stream>>>(T1, A5n, out);
}
